// MultiHeadAttentionL_21852793602475
// MI455X (gfx1250) — compile-verified, matrix operand order checked by reading, not run on hardware
//
#include <hip/hip_runtime.h>

// ---------------------------------------------------------------------------
// MI455X (gfx1250) fused MHA: bf16 WMMA projections + flash attention.
// wave32, V_WMMA_F32_16X16X32_BF16 everywhere (f32 accumulate).
// ---------------------------------------------------------------------------

typedef __attribute__((ext_vector_type(16))) __bf16 v16bf;
typedef __attribute__((ext_vector_type(8)))  float  v8f;

union ABu {
    uint4          u4[2];
    unsigned short s[16];
    v16bf          bf;
};

// float -> bf16 (round-to-nearest-even), software fallback
__device__ __forceinline__ unsigned short f2bf(float f) {
    union { float f; unsigned u; } x; x.f = f;
    unsigned u = x.u;
    unsigned r = (u + 0x7FFFu + ((u >> 16) & 1u)) >> 16;
    return (unsigned short)r;
}

// pack two floats into two bf16 (one dword); hardware v_cvt_pk_bf16_f32 if present
__device__ __forceinline__ unsigned pack2bf(float a, float b) {
#if __has_builtin(__builtin_amdgcn_cvt_pk_bf16_f32)
    auto v = __builtin_amdgcn_cvt_pk_bf16_f32(a, b);
    unsigned u;
    __builtin_memcpy(&u, &v, 4);
    return u;
#else
    return (unsigned)f2bf(a) | ((unsigned)f2bf(b) << 16);
#endif
}

// Load a 16x32 bf16 tile (row-major, row stride `ld` elements) into the
// documented 16-bit A-matrix VGPR layout: lane L holds row L%16; K chunks
// [c0, c0+8) and [c0+16, c0+24) with c0 = (L/16)*8.  B-tiles use the same
// loader on the transposed operand (B kept as N x K).  Works for global or
// LDS-backed pointers (compiler infers the address space).
__device__ __forceinline__ v16bf load_tile_bf16(const unsigned short* base, int ld) {
    int lane = threadIdx.x & 31;
    int r    = lane & 15;
    int c0   = (lane >> 4) << 3;
    const unsigned short* p = base + r * ld + c0;
    ABu u;
    u.u4[0] = *(const uint4*)(p);
    u.u4[1] = *(const uint4*)(p + 16);
    return u.bf;
}

// ---------------------------------------------------------------------------
// Kernel 0: W[256,512] fp32 -> Wt[512,256] bf16 for all four projections.
// ---------------------------------------------------------------------------
__global__ void wt_kernel(const float* __restrict__ Wk, const float* __restrict__ Wv,
                          const float* __restrict__ Wq, const float* __restrict__ Wr,
                          unsigned short* __restrict__ wt) {
    int t   = blockIdx.x * 256 + threadIdx.x;   // 4 * 512 * 256 total
    int p   = t >> 17;
    int rem = t & 131071;
    int n   = rem >> 8;
    int k   = rem & 255;
    const float* W = (p == 0) ? Wk : (p == 1) ? Wv : (p == 2) ? Wq : Wr;
    wt[p * 131072 + n * 256 + k] = f2bf(W[k * 512 + n]);
}

// ---------------------------------------------------------------------------
// Kernel 1: Y = X @ W + b for {K,V,Q,R}.  One 16-row x 512-col strip per
// workgroup; the X strip is converted to bf16 ONCE cooperatively into LDS
// (8x less global fp32 traffic + conversion work), then 8 waves x 4
// col-tiles each run a K loop of 8 x wmma(16x16x32) with A from LDS.
//   proj 0 -> k bf16 flat    proj 1 -> v bf16 transposed per group [g][d][s]
//   proj 2 -> q bf16 flat    proj 3 -> residual fp32 flat
// ---------------------------------------------------------------------------
__global__ void proj_kernel(const float* __restrict__ key0, const float* __restrict__ value0,
                            const float* __restrict__ query0,
                            const float* __restrict__ bk, const float* __restrict__ bv,
                            const float* __restrict__ bq, const float* __restrict__ br,
                            const unsigned short* __restrict__ wt,
                            unsigned short* __restrict__ q16, unsigned short* __restrict__ k16,
                            unsigned short* __restrict__ vt, float* __restrict__ resf) {
    __shared__ __align__(16) unsigned short ldsA[16][256];   // 8 KiB bf16 X strip

    int proj = blockIdx.z;
    int row0 = blockIdx.x * 16;
    int wave = threadIdx.x >> 5;
    int lane = threadIdx.x & 31;
    int n0   = wave * 64;

    const float* X    = (proj == 0) ? key0 : (proj == 1) ? value0 : query0;
    const float* bias = (proj == 0) ? bk : (proj == 1) ? bv : (proj == 2) ? bq : br;
    const unsigned short* Wp = wt + proj * 131072;

    // --- cooperative fp32 -> bf16 staging of the 16x256 X strip -----------
    {
        unsigned* lds32 = (unsigned*)&ldsA[0][0];
#pragma unroll
        for (int i = 0; i < 8; ++i) {
            int e  = threadIdx.x + i * 256;      // pair index 0..2047
            int r  = e >> 7;
            int c2 = (e & 127) << 1;
            const float* p = X + (row0 + r) * 256 + c2;
            lds32[e] = pack2bf(p[0], p[1]);
        }
    }
    __syncthreads();

    v8f acc[4];
#pragma unroll
    for (int t = 0; t < 4; ++t)
#pragma unroll
        for (int i = 0; i < 8; ++i) acc[t][i] = 0.0f;

#pragma unroll
    for (int kt = 0; kt < 8; ++kt) {
        v16bf A = load_tile_bf16(&ldsA[0][0] + kt * 32, 256);   // ds_load_b128 x2
#pragma unroll
        for (int t = 0; t < 4; ++t) {
            v16bf B = load_tile_bf16(Wp + (n0 + t * 16) * 256 + kt * 32, 256);
            acc[t] = __builtin_amdgcn_wmma_f32_16x16x32_bf16(
                false, A, false, B, (short)0, acc[t], false, false);
        }
    }

    int half = lane >> 4;
    int coll = lane & 15;
#pragma unroll
    for (int t = 0; t < 4; ++t) {
        int col  = n0 + t * 16 + coll;
        float bb = bias[col];
#pragma unroll
        for (int r = 0; r < 8; ++r) {
            int   row = row0 + r + 8 * half;
            float val = acc[t][r] + bb;
            int   e   = row * 512 + col;
            if (proj == 3) {
                resf[e] = val;
            } else if (proj == 2) {
                q16[e] = f2bf(val);
            } else if (proj == 0) {
                k16[e] = f2bf(val);
            } else {
                // scatter v into per-group transposed layout vt[g][d'][s']
                int g = e >> 15, rem = e & 32767, sp = rem >> 5, dp = rem & 31;
                vt[(g << 15) + dp * 1024 + sp] = f2bf(val);
            }
        }
    }
}

// ---------------------------------------------------------------------------
// Kernel 2: flash attention per (group g, 128-row Q block).  8 waves, each
// owns 16 Q rows; iterates 1024 keys in 32-wide tiles; online softmax;
// P bounced through per-wave LDS (C-layout -> A-layout) for the PV matmul.
// Next iteration's K/V tiles are prefetched (global_prefetch_b8).
// ---------------------------------------------------------------------------
__global__ void attn_kernel(const unsigned short* __restrict__ q16,
                            const unsigned short* __restrict__ k16,
                            const unsigned short* __restrict__ vt,
                            const float* __restrict__ resf, float* __restrict__ out) {
    __shared__ __align__(16) unsigned short ldsP[8][16][32];

    int qb   = blockIdx.x;        // 0..7
    int g    = blockIdx.y;        // 0..255
    int wave = threadIdx.x >> 5;
    int lane = threadIdx.x & 31;
    int half = lane >> 4;
    int coll = lane & 15;
    int qrow0 = qb * 128 + wave * 16;   // group-local row base for this wave

    const unsigned short* qg = q16 + (g << 15);
    const unsigned short* kg = k16 + (g << 15);
    const unsigned short* vg = vt  + (g << 15);

    v16bf Qa = load_tile_bf16(qg + qrow0 * 32, 32);

    v8f  acc0, acc1;
    float m[8], l[8];
#pragma unroll
    for (int r = 0; r < 8; ++r) { acc0[r] = 0.f; acc1[r] = 0.f; m[r] = -3.0e38f; l[r] = 0.f; }

    const float scale = 0.70710678118f;   // (DK // H) ** -0.5 == 2^-0.5

    for (int sk0 = 0; sk0 < 1024; sk0 += 32) {
        // prefetch next iteration's K tile-pair (2 KiB: 64 B per lane) and
        // the V rows' next 64 B segment -> global_prefetch_b8
        if (sk0 + 32 < 1024) {
            __builtin_prefetch(kg + (sk0 + 32) * 32 + lane * 32, 0, 0);
            __builtin_prefetch(vg + lane * 1024 + (sk0 + 32), 0, 0);
        }

        v16bf Kb0 = load_tile_bf16(kg + sk0 * 32, 32);
        v16bf Kb1 = load_tile_bf16(kg + (sk0 + 16) * 32, 32);
        v8f z;
#pragma unroll
        for (int r = 0; r < 8; ++r) z[r] = 0.f;
        v8f s0 = __builtin_amdgcn_wmma_f32_16x16x32_bf16(false, Qa, false, Kb0, (short)0, z, false, false);
        v8f s1 = __builtin_amdgcn_wmma_f32_16x16x32_bf16(false, Qa, false, Kb1, (short)0, z, false, false);

        float p0[8], p1[8];
#pragma unroll
        for (int r = 0; r < 8; ++r) {
            float a = s0[r] * scale;
            float b = s1[r] * scale;
            // row max across the 16 lanes holding this row (cols of both tiles)
            float v = fmaxf(a, b);
            v = fmaxf(v, __shfl_xor(v, 1, 16));
            v = fmaxf(v, __shfl_xor(v, 2, 16));
            v = fmaxf(v, __shfl_xor(v, 4, 16));
            v = fmaxf(v, __shfl_xor(v, 8, 16));
            float mn    = fmaxf(m[r], v);
            float alpha = __expf(m[r] - mn);
            m[r] = mn;
            p0[r] = __expf(a - mn);
            p1[r] = __expf(b - mn);
            float rs = p0[r] + p1[r];
            rs += __shfl_xor(rs, 1, 16);
            rs += __shfl_xor(rs, 2, 16);
            rs += __shfl_xor(rs, 4, 16);
            rs += __shfl_xor(rs, 8, 16);
            l[r]    = l[r] * alpha + rs;
            acc0[r] *= alpha;
            acc1[r] *= alpha;
        }

        // P: C layout (lane=col, vgpr=row) -> LDS -> A layout (lane=row, chunks of K)
#pragma unroll
        for (int r = 0; r < 8; ++r) {
            int row = r + 8 * half;
            ldsP[wave][row][coll]      = f2bf(p0[r]);
            ldsP[wave][row][coll + 16] = f2bf(p1[r]);
        }
        asm volatile("s_wait_dscnt 0" ::: "memory");   // wave-local; LDS is in-order per wave
        v16bf Pa;
        {
            int r  = lane & 15;
            int c0 = (lane >> 4) << 3;
            const unsigned short* p = &ldsP[wave][r][c0];
            ABu u;
            u.u4[0] = *(const uint4*)(p);
            u.u4[1] = *(const uint4*)(p + 16);
            Pa = u.bf;
        }

        v16bf Vb0 = load_tile_bf16(vg + 0  * 1024 + sk0, 1024);
        v16bf Vb1 = load_tile_bf16(vg + 16 * 1024 + sk0, 1024);
        acc0 = __builtin_amdgcn_wmma_f32_16x16x32_bf16(false, Pa, false, Vb0, (short)0, acc0, false, false);
        acc1 = __builtin_amdgcn_wmma_f32_16x16x32_bf16(false, Pa, false, Vb1, (short)0, acc1, false, false);
    }

    // epilogue: context / l + residual, relu, fp32 store
#pragma unroll
    for (int r = 0; r < 8; ++r) {
        float inv = 1.0f / l[r];
        int   row = qrow0 + r + 8 * half;
        int   e0  = (g << 15) + row * 32 + coll;
        int   e1  = e0 + 16;
        float v0  = resf[e0] + acc0[r] * inv;
        float v1  = resf[e1] + acc1[r] * inv;
        out[e0] = fmaxf(v0, 0.0f);
        out[e1] = fmaxf(v1, 0.0f);
    }
}

// ---------------------------------------------------------------------------
extern "C" void kernel_launch(void* const* d_in, const int* in_sizes, int n_in,
                              void* d_out, int out_size, void* d_ws, size_t ws_size,
                              hipStream_t stream) {
    const float* key0   = (const float*)d_in[0];
    const float* value0 = (const float*)d_in[1];
    const float* query0 = (const float*)d_in[2];
    const float* Wk = (const float*)d_in[3];
    const float* bk = (const float*)d_in[4];
    const float* Wv = (const float*)d_in[5];
    const float* bv = (const float*)d_in[6];
    const float* Wq = (const float*)d_in[7];
    const float* bq = (const float*)d_in[8];
    const float* Wr = (const float*)d_in[9];
    const float* br = (const float*)d_in[10];

    // workspace layout (bytes):
    //   wt  bf16: 4*512*256*2      = 1 MiB
    //   q16 bf16: 16384*512*2      = 16 MiB
    //   k16 bf16: 16384*512*2      = 16 MiB
    //   vt  bf16: 16384*512*2      = 16 MiB  (per-group transposed)
    //   res f32 : 16384*512*4      = 32 MiB
    char* ws = (char*)d_ws;
    unsigned short* wt  = (unsigned short*)(ws);
    unsigned short* q16 = (unsigned short*)(ws + 1048576);
    unsigned short* k16 = (unsigned short*)(ws + 1048576 + 1u * 16777216u);
    unsigned short* vt  = (unsigned short*)(ws + 1048576 + 2u * 16777216u);
    float*          resf = (float*)(ws + 1048576 + 3u * 16777216u);
    float* outp = (float*)d_out;

    wt_kernel<<<2048, 256, 0, stream>>>(Wk, Wv, Wq, Wr, wt);

    dim3 g1(1024, 1, 4);   // 1024 row strips x 4 projections
    proj_kernel<<<g1, 256, 0, stream>>>(key0, value0, query0, bk, bv, bq, br,
                                        wt, q16, k16, vt, resf);

    dim3 g2(8, 256);       // 8 Q blocks x 256 groups
    attn_kernel<<<g2, 256, 0, stream>>>(q16, k16, vt, resf, outp);
}
